// FlexibleLQmul_48713519071897
// MI455X (gfx1250) — matrix operand order repaired and checked by reading, not compiled, not run on hardware
//
#include <hip/hip_runtime.h>

#define BB   16
#define JJ   2048
#define HH   2048
#define OUTL 128
#define HD   16

typedef __bf16    bf16;
typedef _Float16  f16;
typedef __attribute__((ext_vector_type(16))) __bf16   bf16x16;
typedef __attribute__((ext_vector_type(8)))  __bf16   bf16x8;
typedef __attribute__((ext_vector_type(4)))  __bf16   bf16x4;
typedef __attribute__((ext_vector_type(16))) _Float16 f16x16;
typedef __attribute__((ext_vector_type(8)))  _Float16 f16x8;
typedef __attribute__((ext_vector_type(8)))  float    f32x8;
typedef __attribute__((ext_vector_type(4)))  float    f32x4;

// ---- A-operand fragment helpers (16-bit A 16x32: K = k0+[0..8) U k0+[16..24)) ----
struct Frag32 { f32x4 a0, a1, b0, b1; };

__device__ __forceinline__ Frag32 ldfragA_f32(const float* p) {
  Frag32 f;
  f.a0 = *(const f32x4*)(p);
  f.a1 = *(const f32x4*)(p + 4);
  f.b0 = *(const f32x4*)(p + 16);
  f.b1 = *(const f32x4*)(p + 20);
  return f;
}

__device__ __forceinline__ bf16x16 fragA_to_bf16(Frag32 f) {
  bf16x16 r;
#pragma unroll
  for (int i = 0; i < 4; ++i) {
    r[i]      = (bf16)f.a0[i];
    r[i + 4]  = (bf16)f.a1[i];
    r[i + 8]  = (bf16)f.b0[i];
    r[i + 12] = (bf16)f.b1[i];
  }
  return r;
}

__device__ __forceinline__ float fragA_dot(Frag32 a, Frag32 b) {
  float s = 0.f;
#pragma unroll
  for (int i = 0; i < 4; ++i)
    s += a.a0[i]*b.a0[i] + a.a1[i]*b.a1[i] + a.b0[i]*b.b0[i] + a.b1[i]*b.b1[i];
  return s;
}

__device__ __forceinline__ f32x8 wmma_bf16(bf16x16 a, bf16x16 b, f32x8 c) {
  return __builtin_amdgcn_wmma_f32_16x16x32_bf16(false, a, false, b, (short)0, c, false, false);
}
__device__ __forceinline__ f32x8 wmma_f16(f16x16 a, f16x16 b, f32x8 c) {
  return __builtin_amdgcn_wmma_f32_16x16x32_f16(false, a, false, b, (short)0, c, false, false);
}

// ---------------------------------------------------------------------------
// Prep kernels
// ---------------------------------------------------------------------------
// wsum[c] = sum_h wk_w[h][c]  (coalesced column sums, 16 MB read once)
__global__ __launch_bounds__(256) void k_wsum(const float* __restrict__ wk_w,
                                              float* __restrict__ wsum) {
  int c = blockIdx.x * 256 + threadIdx.x;   // 2048 columns, grid = 8
  float s = 0.f;
  for (int h = 0; h < HH; ++h) s += wk_w[(size_t)h * HH + c];
  wsum[c] = s;
}

__global__ __launch_bounds__(256) void k_bsum(const float* __restrict__ wk_b,
                                              float* __restrict__ bsum) {
  __shared__ float sh[256];
  int t = threadIdx.x;
  float s = 0.f;
  for (int i = t; i < HH; i += 256) s += wk_b[i];
  sh[t] = s;
  __syncthreads();
  for (int o = 128; o > 0; o >>= 1) { if (t < o) sh[t] += sh[t + o]; __syncthreads(); }
  if (t == 0) bsum[0] = sh[0];
}

// Pre-convert a weight matrix f32 -> bf16 (removes per-iteration B-side cvts
// from the hot projection loop; weights are reused 2048x from L2)
__global__ __launch_bounds__(256) void k_cvt_bf16(const float* __restrict__ in,
                                                  bf16* __restrict__ out) {
  int idx = (blockIdx.x * 256 + threadIdx.x) * 4;
  f32x4 x = *(const f32x4*)(in + idx);
  bf16x4 y;
#pragma unroll
  for (int i = 0; i < 4; ++i) y[i] = (bf16)x[i];
  *(bf16x4*)(out + idx) = y;
}

// ---------------------------------------------------------------------------
// Fused projection: ONE streaming pass over vec (256 MB -> HBM-bound ~11us).
// Grid: (b, j-tile of 16) = 2048 workgroups, 9 waves each.
//   waves 0..7 : q i-tile w   -> qT[b][i][j] bf16 (transposed, contiguous j)
//   wave  8    : v (WMMA)     -> vT[b][d][j] f16
//                S (VALU dot) -> S[b][j]     f32   (S = vec.wsum + bsum)
// B operands (weights) are pre-converted bf16: two 16B fragment loads each.
// ---------------------------------------------------------------------------
__global__ __launch_bounds__(288) void k_proj(const float* __restrict__ vec,
                                              const bf16* __restrict__ wqh,
                                              const float* __restrict__ wq_b,
                                              const bf16* __restrict__ wvh,
                                              const float* __restrict__ wv_b,
                                              const float* __restrict__ wsum,
                                              const float* __restrict__ bsum,
                                              bf16* __restrict__ qT,
                                              f16*  __restrict__ vT,
                                              float* __restrict__ S) {
  const int wg   = blockIdx.x;
  const int b    = wg >> 7;
  const int jb   = wg & 127;
  const int tid  = threadIdx.x;
  const int lane = tid & 31;
  const int w    = tid >> 5;            // 0..8
  const int j0   = jb * 16;
  const int lrow = lane & 15;
  const int k0   = (lane >> 4) * 8;     // A-fragment K base (split pattern)
  const int kb0  = (lane >> 4) * 8;     // B-fragment K base (split pattern, same as A)
  const int r0   = k0;                  // D-fragment row base

  const float* arow = vec + ((size_t)b * JJ + (j0 + lrow)) * HH + k0;

  if (w < 8) {
    // ---- q projection tile: rows j (16), cols i0..i0+15 ----
    const int i0 = w * 16;
    const bf16* brow = wqh + (size_t)(i0 + lrow) * HH + kb0;
    f32x8 acc = {};
    for (int c0 = 0; c0 < HH; c0 += 32) {
      __builtin_prefetch(arow + c0 + 64, 0, 1);   // global_prefetch_b8
      bf16x16 bm = __builtin_shufflevector(*(const bf16x8*)(brow + c0), *(const bf16x8*)(brow + c0 + 16), 0, 1, 2, 3, 4, 5, 6, 7, 8, 9, 10, 11, 12, 13, 14, 15);  // split K, matches A
      acc = wmma_bf16(fragA_to_bf16(ldfragA_f32(arow + c0)), bm, acc);
    }
    const float bias = wq_b[i0 + lrow];           // D col = i
    bf16x8 o;
#pragma unroll
    for (int r = 0; r < 8; ++r) o[r] = (bf16)(acc[r] + bias);
    // qT[b][i0+lrow][j0+r0 .. +8)  -- contiguous j, one b128 store
    *(bf16x8*)(qT + ((size_t)(b * OUTL + i0 + lrow)) * JJ + j0 + r0) = o;
  } else {
    // ---- v projection (WMMA) + S row-sums (VALU) ----
    const bf16* brow = wvh + (size_t)lrow * HH + kb0;
    f32x8 acc = {};
    float sacc = 0.f;
    for (int c0 = 0; c0 < HH; c0 += 32) {
      Frag32 a = ldfragA_f32(arow + c0);
      sacc    += fragA_dot(a, ldfragA_f32(wsum + c0 + k0));
      bf16x16 bm = __builtin_shufflevector(*(const bf16x8*)(brow + c0), *(const bf16x8*)(brow + c0 + 16), 0, 1, 2, 3, 4, 5, 6, 7, 8, 9, 10, 11, 12, 13, 14, 15);
      acc      = wmma_bf16(fragA_to_bf16(a), bm, acc);
    }
    const float vb = wv_b[lrow];                  // D col = d
    f16x8 o;
#pragma unroll
    for (int r = 0; r < 8; ++r) o[r] = (f16)(acc[r] + vb);
    // vT[b][d][j0+r0 .. +8) -- contiguous j
    *(f16x8*)(vT + ((size_t)(b * HD + lrow)) * JJ + j0 + r0) = o;
    // lane l and l+16 cover complementary c-slices of row j0+lrow
    sacc += __shfl_xor(sacc, 16, 32);
    if (lane < 16) S[(size_t)b * JJ + j0 + lrow] = sacc + bsum[0];
  }
}

// ---------------------------------------------------------------------------
// Fused attention: attn[b,i,j] = qT[b][i][j] * S[b][j]; e = exp(tanh(.))
// (tanh output bounded => no max-subtraction needed); online softmax:
//   ret[b,i,d] = (sum_j e * v[j,d]) / (sum_j e)    via f16 WMMA P.V
// Grid: (b, i-block of 16) = 128 workgroups, 8 waves split the j range.
// ---------------------------------------------------------------------------
__global__ __launch_bounds__(256) void k_attn(const bf16* __restrict__ qT,
                                              const float* __restrict__ S,
                                              const f16*  __restrict__ vT,
                                              float* __restrict__ out) {
  const int wg   = blockIdx.x;
  const int b    = wg >> 3;
  const int ib   = wg & 7;
  const int tid  = threadIdx.x;
  const int lane = tid & 31;
  const int w    = tid >> 5;            // 0..7
  const int i0   = ib * 16;
  const int lrow = lane & 15;
  const int k0   = (lane >> 4) * 8;
  const int kb0  = (lane >> 4) * 8;
  const int r0   = k0;

  __shared__ float red_num[8][16][16];
  __shared__ float red_den[8][16];

  const bf16* qrow  = qT + ((size_t)(b * OUTL + i0 + lrow)) * JJ + k0;  // A: row = i
  const float* srow = S + (size_t)b * JJ + k0;
  const f16*  vrow  = vT + ((size_t)(b * HD + lrow)) * JJ + kb0;        // B: col = d

  f32x8 num = {};
  float den = 0.f;

  for (int jc = w; jc < JJ / 32; jc += 8) {
    const int j0 = jc * 32;
    // Build E A-fragment in registers: e = exp(tanh(q * S))
    bf16x8 qlo = *(const bf16x8*)(qrow + j0);
    bf16x8 qhi = *(const bf16x8*)(qrow + j0 + 16);
    Frag32 sf  = ldfragA_f32(srow + j0);
    f16x16 E;
#pragma unroll
    for (int e = 0; e < 8; ++e) {
      float sv = (e < 4) ? sf.a0[e] : sf.a1[e - 4];
      float x  = __expf(tanhf((float)qlo[e] * sv));
      den += x; E[e] = (f16)x;
    }
#pragma unroll
    for (int e = 0; e < 8; ++e) {
      float sv = (e < 4) ? sf.b0[e] : sf.b1[e - 4];
      float x  = __expf(tanhf((float)qhi[e] * sv));
      den += x; E[8 + e] = (f16)x;
    }
    f16x16 V = __builtin_shufflevector(*(const f16x8*)(vrow + j0), *(const f16x8*)(vrow + j0 + 16), 0, 1, 2, 3, 4, 5, 6, 7, 8, 9, 10, 11, 12, 13, 14, 15);  // split K, matches A
    num = wmma_f16(E, V, num);
  }

  // den: lane pair (l, l+16) jointly covers the wave's full j slices of row lrow
  den += __shfl_xor(den, 16, 32);
  if (lane < 16) red_den[w][lrow] = den;
#pragma unroll
  for (int r = 0; r < 8; ++r) red_num[w][r0 + r][lrow] = num[r];  // D: col=d, rows r0+r
  __syncthreads();

  // Cross-wave reduce + normalize: 256 threads = 16 rows x 16 d
  const int row = tid >> 4, d = tid & 15;
  float ns = 0.f, ds = 0.f;
#pragma unroll
  for (int ww = 0; ww < 8; ++ww) { ns += red_num[ww][row][d]; ds += red_den[ww][row]; }
  out[((size_t)(b * OUTL + i0 + row)) * HD + d] = ns / ds;
}

// ---------------------------------------------------------------------------
extern "C" void kernel_launch(void* const* d_in, const int* in_sizes, int n_in,
                              void* d_out, int out_size, void* d_ws, size_t ws_size,
                              hipStream_t stream) {
  const float* vec  = (const float*)d_in[0];
  const float* wq_w = (const float*)d_in[1];
  const float* wq_b = (const float*)d_in[2];
  const float* wk_w = (const float*)d_in[3];
  const float* wk_b = (const float*)d_in[4];
  const float* wv_w = (const float*)d_in[5];
  const float* wv_b = (const float*)d_in[6];
  float* out = (float*)d_out;

  // Workspace carve-out (~10.6 MB total)
  char* ws = (char*)d_ws;
  bf16*  qT   = (bf16*)(ws);                               // 16*128*2048*2 = 8 MB
  f16*   vT   = (f16*)(ws + (8u << 20));                   // 16*16*2048*2  = 1 MB
  float* Sbuf = (float*)(ws + (9u << 20));                 // 16*2048*4     = 128 KB
  float* wsum = (float*)(ws + (9u << 20) + (256u << 10));  // 8 KB
  float* bsum = (float*)(ws + (9u << 20) + (264u << 10));  // 4 B
  bf16*  wqh  = (bf16*)(ws + (10u << 20));                 // 128*2048*2 = 512 KB
  bf16*  wvh  = (bf16*)(ws + (10u << 20) + (512u << 10));  // 16*2048*2  = 64 KB

  k_wsum<<<dim3(HH / 256), dim3(256), 0, stream>>>(wk_w, wsum);
  k_bsum<<<dim3(1), dim3(256), 0, stream>>>(wk_b, bsum);
  k_cvt_bf16<<<dim3((OUTL * HH) / 1024), dim3(256), 0, stream>>>(wq_w, wqh);
  k_cvt_bf16<<<dim3((HD * HH) / 1024), dim3(256), 0, stream>>>(wv_w, wvh);
  k_proj<<<dim3(BB * (JJ / 16)), dim3(288), 0, stream>>>(vec, wqh, wq_b, wvh, wv_b,
                                                         wsum, bsum, qT, vT, Sbuf);
  k_attn<<<dim3(BB * (OUTL / 16)), dim3(256), 0, stream>>>(qT, Sbuf, vT, out);
}
